// Net_75230647157015
// MI455X (gfx1250) — matrix operand order repaired and checked by reading, not compiled, not run on hardware
//
#include <hip/hip_runtime.h>
#include <hip/hip_bf16.h>

typedef __attribute__((ext_vector_type(16))) _Float16 v16h;
typedef __attribute__((ext_vector_type(8)))  float    v8f;

union FragU { v16h v; uint4 q[2]; };

// Workspace layout (halves) for transposed f16 weights [N][K] (batch-major)
#define OFF_W1   0        // [128][256]
#define OFF_W2   32768    // [64][128]
#define OFF_EW1  40960    // 4 x [128][64]
#define OFF_EW2  73728    // 4 x [64][128]
#define OFF_GW1  106496   // [32][64]
#define OFF_SW   108544   // [64][64]
#define OFF_HW1  112640   // 3 x [32][64]
#define WS_HALVES 118784

#define SMEM_BYTES 117760

__device__ __forceinline__ float silu_f(float x) {
    return x / (1.0f + __expf(-x));
}

__device__ __forceinline__ v8f wmma16(v16h a, v16h b, v8f c) {
    return __builtin_amdgcn_wmma_f32_16x16x32_f16(false, a, false, b, (short)0, c, false, false);
}

// B fragment: wT is [N][Kld] f16 row-major (k contiguous). lane: n = n0 + (lane%16),
// halves i -> k = k0 + (i>>3)*16 + (lane/16)*8 + (i&7)  => two 16B runs.
__device__ __forceinline__ v16h b_frag(const _Float16* __restrict__ wT, int Kld,
                                       int n0, int k0, int m16, int hi) {
    const _Float16* p = wT + (size_t)(n0 + m16) * Kld + k0 + hi * 8;
    FragU f;
    f.q[0] = ((const uint4*)p)[0];
    f.q[1] = ((const uint4*)p)[2];
    return f.v;
}

// A fragment from LDS f16 buffer [16][Kld]: lane m=lane%16; halves i:
// k = k0 + (i>>3)*16 + hi*8 + (i&7)  => two contiguous 8-half (16B) runs.
__device__ __forceinline__ v16h a_frag_lds(const _Float16* buf, int Kld,
                                           int k0, int m16, int hi) {
    const _Float16* p = buf + m16 * Kld + k0 + hi * 8;
    FragU f;
    f.q[0] = *(const uint4*)(p);
    f.q[1] = *(const uint4*)(p + 16);
    return f.v;
}

// A fragment from global f32 row (convert to f16). p0 points at the i=0..7 run.
__device__ __forceinline__ v16h a_frag_g32(const float* __restrict__ p0) {
    float4 r0 = ((const float4*)p0)[0];
    float4 r1 = ((const float4*)p0)[1];
    float4 r2 = ((const float4*)(p0 + 16))[0];
    float4 r3 = ((const float4*)(p0 + 16))[1];
    v16h a;
    a[0]=(_Float16)r0.x;  a[1]=(_Float16)r0.y;  a[2]=(_Float16)r0.z;  a[3]=(_Float16)r0.w;
    a[4]=(_Float16)r1.x;  a[5]=(_Float16)r1.y;  a[6]=(_Float16)r1.z;  a[7]=(_Float16)r1.w;
    a[8]=(_Float16)r2.x;  a[9]=(_Float16)r2.y;  a[10]=(_Float16)r2.z; a[11]=(_Float16)r2.w;
    a[12]=(_Float16)r3.x; a[13]=(_Float16)r3.y; a[14]=(_Float16)r3.z; a[15]=(_Float16)r3.w;
    return a;
}

// ---- weight prep: f32 [b][K][N] -> f16 [b][N][K] ----
__global__ void transpose_f16_kernel(const float* __restrict__ src, _Float16* __restrict__ dst,
                                     int K, int N, int total) {
    int idx = blockIdx.x * blockDim.x + threadIdx.x;
    if (idx >= total) return;
    int kn = K * N;
    int b = idx / kn;
    int rem = idx - b * kn;
    int k = rem / N;
    int n = rem - k * N;
    dst[(size_t)b * kn + (size_t)n * K + k] = (_Float16)src[idx];
}

// ---- fused MoE net: 256 threads = 8 waves, 16 rows per wave, 128 rows per block ----
__global__ __launch_bounds__(256) void moe_fused_kernel(
    const float* __restrict__ x,
    const float* __restrict__ b1,  const float* __restrict__ lng, const float* __restrict__ lnb,
    const float* __restrict__ b2,
    const float* __restrict__ eb1, const float* __restrict__ eb2,
    const float* __restrict__ gb1, const float* __restrict__ gw2, const float* __restrict__ gb2,
    const float* __restrict__ sb,
    const float* __restrict__ hb1, const float* __restrict__ hw2, const float* __restrict__ hb2,
    const _Float16* __restrict__ ws,
    float* __restrict__ out)
{
    extern __shared__ char smem_raw[];
    const int tid  = threadIdx.x;
    const int lane = tid & 31;
    const int wave = tid >> 5;
    const int m16  = lane & 15;
    const int hi   = lane >> 4;
    const int row0 = blockIdx.x * 128 + wave * 16;

    // per-wave LDS regions
    float*    h1f_w  = (float*)smem_raw + wave * 2048;                 // [16][128] f32 (also g1/hh scratch)
    _Float16* abuf_w = (_Float16*)(smem_raw + 65536) + wave * 2048;    // [16][128] f16
    _Float16* hbuf_w = (_Float16*)(smem_raw + 98304) + wave * 1024;    // [16][64]  f16
    float*    mu_w   = (float*)(smem_raw + 114688) + wave * 32;        // [16] mu, [16] rsig
    float*    gw_w   = (float*)(smem_raw + 115712) + wave * 64;        // [16][4] gate weights

    const v8f Z8 = {0.f,0.f,0.f,0.f,0.f,0.f,0.f,0.f};

    // ---- stage 1: h1 = x @ W1 + b1  (K=256, N=128) ----
    const float* xr = x + (size_t)(row0 + m16) * 256;
    v16h ax[8];
    #pragma unroll
    for (int kt = 0; kt < 8; ++kt) ax[kt] = a_frag_g32(xr + kt * 32 + hi * 8);

    #pragma unroll
    for (int nt = 0; nt < 8; ++nt) {
        // batch all B-fragment loads first (clustered clause, overlaps with WMMA chain)
        v16h bf[8];
        #pragma unroll
        for (int kt = 0; kt < 8; ++kt)
            bf[kt] = b_frag(ws + OFF_W1, 256, nt * 16, kt * 32, m16, hi);
        v8f acc0 = Z8, acc1 = Z8;
        #pragma unroll
        for (int kt = 0; kt < 8; kt += 2) {
            acc0 = wmma16(ax[kt],     bf[kt],     acc0);
            acc1 = wmma16(ax[kt + 1], bf[kt + 1], acc1);
        }
        v8f acc = acc0 + acc1;
        int col = nt * 16 + m16;
        float bb = b1[col];
        #pragma unroll
        for (int r = 0; r < 8; ++r)
            h1f_w[(hi * 8 + r) * 128 + col] = acc[r] + bb;
    }
    __syncthreads();

    // ---- LayerNorm(128) + SiLU -> abuf (f16) ----
    if (lane < 16) {
        const float* rp = h1f_w + lane * 128;
        float s = 0.f, s2 = 0.f;
        #pragma unroll 8
        for (int c = 0; c < 128; ++c) { float v = rp[c]; s += v; s2 += v * v; }
        float mu  = s * (1.0f / 128.0f);
        float var = s2 * (1.0f / 128.0f) - mu * mu;
        mu_w[lane]      = mu;
        mu_w[16 + lane] = rsqrtf(var + 1e-5f);
    }
    __syncthreads();
    for (int idx = lane; idx < 2048; idx += 32) {
        int row = idx >> 7, col = idx & 127;
        float v = h1f_w[idx];
        v = (v - mu_w[row]) * mu_w[16 + row] * lng[col] + lnb[col];
        abuf_w[idx] = (_Float16)silu_f(v);
    }
    __syncthreads();

    // ---- stage 2: h = silu(h1 @ W2 + b2)  (K=128, N=64) -> hbuf ----
    {
        v16h a2[4];
        #pragma unroll
        for (int kt = 0; kt < 4; ++kt) a2[kt] = a_frag_lds(abuf_w, 128, kt * 32, m16, hi);
        #pragma unroll
        for (int nt = 0; nt < 4; ++nt) {
            v16h bf[4];
            #pragma unroll
            for (int kt = 0; kt < 4; ++kt)
                bf[kt] = b_frag(ws + OFF_W2, 128, nt * 16, kt * 32, m16, hi);
            v8f acc0 = Z8, acc1 = Z8;
            acc0 = wmma16(a2[0], bf[0], acc0);
            acc1 = wmma16(a2[1], bf[1], acc1);
            acc0 = wmma16(a2[2], bf[2], acc0);
            acc1 = wmma16(a2[3], bf[3], acc1);
            v8f acc = acc0 + acc1;
            int col = nt * 16 + m16;
            float bb = b2[col];
            #pragma unroll
            for (int r = 0; r < 8; ++r)
                hbuf_w[(hi * 8 + r) * 64 + col] = (_Float16)silu_f(acc[r] + bb);
        }
    }
    __syncthreads();

    // ---- gate: g1 = silu(h @ GW1 + gb1); logits = g1 @ GW2 + gb2; softmax ----
    v16h a_h0 = a_frag_lds(hbuf_w, 64, 0,  m16, hi);
    v16h a_h1 = a_frag_lds(hbuf_w, 64, 32, m16, hi);
    {
        float* g1 = h1f_w; // reuse as [16][32] f32
        #pragma unroll
        for (int nt = 0; nt < 2; ++nt) {
            v16h bf0 = b_frag(ws + OFF_GW1, 64, nt * 16, 0,  m16, hi);
            v16h bf1 = b_frag(ws + OFF_GW1, 64, nt * 16, 32, m16, hi);
            v8f acc0 = wmma16(a_h0, bf0, Z8);
            v8f acc1 = wmma16(a_h1, bf1, Z8);
            v8f acc = acc0 + acc1;
            int col = nt * 16 + m16;
            float bb = gb1[col];
            #pragma unroll
            for (int r = 0; r < 8; ++r)
                g1[(hi * 8 + r) * 32 + col] = silu_f(acc[r] + bb);
        }
        __syncthreads();
        if (lane < 16) {
            float z[4];
            #pragma unroll
            for (int e = 0; e < 4; ++e) {
                float t = gb2[e];
                #pragma unroll 8
                for (int j = 0; j < 32; ++j) t += g1[lane * 32 + j] * gw2[j * 4 + e];
                z[e] = t;
            }
            float mx = fmaxf(fmaxf(z[0], z[1]), fmaxf(z[2], z[3]));
            float se = 0.f;
            #pragma unroll
            for (int e = 0; e < 4; ++e) { z[e] = __expf(z[e] - mx); se += z[e]; }
            float inv = 1.0f / se;
            #pragma unroll
            for (int e = 0; e < 4; ++e) gw_w[lane * 4 + e] = z[e] * inv;
        }
        __syncthreads();
    }

    // ---- experts: moe = sum_e gate_w[e] * silu(silu(h@EW1[e]+eb1)@EW2[e]+eb2) ----
    float moe[4][8];
    #pragma unroll
    for (int nt = 0; nt < 4; ++nt)
        #pragma unroll
        for (int r = 0; r < 8; ++r) moe[nt][r] = 0.f;

    for (int e = 0; e < 4; ++e) {
        const _Float16* w1e = ws + OFF_EW1 + e * 8192;  // [128][64]
        #pragma unroll
        for (int nt = 0; nt < 8; ++nt) {
            v16h bf0 = b_frag(w1e, 64, nt * 16, 0,  m16, hi);
            v16h bf1 = b_frag(w1e, 64, nt * 16, 32, m16, hi);
            v8f acc0 = wmma16(a_h0, bf0, Z8);
            v8f acc1 = wmma16(a_h1, bf1, Z8);
            v8f acc = acc0 + acc1;
            int col = nt * 16 + m16;
            float bb = eb1[e * 128 + col];
            #pragma unroll
            for (int r = 0; r < 8; ++r)
                abuf_w[(hi * 8 + r) * 128 + col] = (_Float16)silu_f(acc[r] + bb);
        }
        __syncthreads();

        v16h ae[4];
        #pragma unroll
        for (int kt = 0; kt < 4; ++kt) ae[kt] = a_frag_lds(abuf_w, 128, kt * 32, m16, hi);
        float gwr[8];
        #pragma unroll
        for (int r = 0; r < 8; ++r) gwr[r] = gw_w[(hi * 8 + r) * 4 + e];

        const _Float16* w2e = ws + OFF_EW2 + e * 8192;  // [64][128]
        #pragma unroll
        for (int nt = 0; nt < 4; ++nt) {
            v16h bf[4];
            #pragma unroll
            for (int kt = 0; kt < 4; ++kt)
                bf[kt] = b_frag(w2e, 128, nt * 16, kt * 32, m16, hi);
            v8f acc0 = Z8, acc1 = Z8;
            acc0 = wmma16(ae[0], bf[0], acc0);
            acc1 = wmma16(ae[1], bf[1], acc1);
            acc0 = wmma16(ae[2], bf[2], acc0);
            acc1 = wmma16(ae[3], bf[3], acc1);
            v8f acc = acc0 + acc1;
            int col = nt * 16 + m16;
            float bb = eb2[e * 64 + col];
            #pragma unroll
            for (int r = 0; r < 8; ++r)
                moe[nt][r] += gwr[r] * silu_f(acc[r] + bb);
        }
        __syncthreads();
    }

    // ---- shared trunk: feat = silu(moe @ SW + sb) ----
    #pragma unroll
    for (int nt = 0; nt < 4; ++nt) {
        int col = nt * 16 + m16;
        #pragma unroll
        for (int r = 0; r < 8; ++r)
            hbuf_w[(hi * 8 + r) * 64 + col] = (_Float16)moe[nt][r];
    }
    __syncthreads();
    {
        v16h am0 = a_frag_lds(hbuf_w, 64, 0,  m16, hi);
        v16h am1 = a_frag_lds(hbuf_w, 64, 32, m16, hi);
        _Float16* fbuf = abuf_w; // [16][64] f16
        #pragma unroll
        for (int nt = 0; nt < 4; ++nt) {
            v16h bf0 = b_frag(ws + OFF_SW, 64, nt * 16, 0,  m16, hi);
            v16h bf1 = b_frag(ws + OFF_SW, 64, nt * 16, 32, m16, hi);
            v8f acc0 = wmma16(am0, bf0, Z8);
            v8f acc1 = wmma16(am1, bf1, Z8);
            v8f acc = acc0 + acc1;
            int col = nt * 16 + m16;
            float bb = sb[col];
            #pragma unroll
            for (int r = 0; r < 8; ++r)
                fbuf[(hi * 8 + r) * 64 + col] = (_Float16)silu_f(acc[r] + bb);
        }
    }
    __syncthreads();

    // ---- heads: out[:,k] = silu(feat @ HW1[k] + hb1) @ HW2[k] + hb2 ----
    v16h af0 = a_frag_lds(abuf_w, 64, 0,  m16, hi);
    v16h af1 = a_frag_lds(abuf_w, 64, 32, m16, hi);
    float* hh = h1f_w; // reuse as [16][32] f32
    #pragma unroll
    for (int k3 = 0; k3 < 3; ++k3) {
        const _Float16* wh = ws + OFF_HW1 + k3 * 2048; // [32][64]
        #pragma unroll
        for (int nt = 0; nt < 2; ++nt) {
            v16h bf0 = b_frag(wh, 64, nt * 16, 0,  m16, hi);
            v16h bf1 = b_frag(wh, 64, nt * 16, 32, m16, hi);
            v8f acc0 = wmma16(af0, bf0, Z8);
            v8f acc1 = wmma16(af1, bf1, Z8);
            v8f acc = acc0 + acc1;
            int col = nt * 16 + m16;
            float bb = hb1[k3 * 32 + col];
            #pragma unroll
            for (int r = 0; r < 8; ++r)
                hh[(hi * 8 + r) * 32 + col] = silu_f(acc[r] + bb);
        }
        __syncthreads();
        if (lane < 16) {
            float o = hb2[k3];
            #pragma unroll 8
            for (int j = 0; j < 32; ++j) o += hh[lane * 32 + j] * hw2[k3 * 32 + j];
            out[(size_t)(row0 + lane) * 3 + k3] = o;
        }
        __syncthreads();
    }
}

extern "C" void kernel_launch(void* const* d_in, const int* in_sizes, int n_in,
                              void* d_out, int out_size, void* d_ws, size_t ws_size,
                              hipStream_t stream) {
    (void)n_in; (void)out_size; (void)ws_size;
    const float* x   = (const float*)d_in[0];
    const float* b1  = (const float*)d_in[2];
    const float* lng = (const float*)d_in[3];
    const float* lnb = (const float*)d_in[4];
    const float* b2  = (const float*)d_in[6];
    const float* eb1 = (const float*)d_in[8];
    const float* eb2 = (const float*)d_in[10];
    const float* gb1 = (const float*)d_in[12];
    const float* gw2 = (const float*)d_in[13];
    const float* gb2 = (const float*)d_in[14];
    const float* sb  = (const float*)d_in[16];
    const float* hb1 = (const float*)d_in[18];
    const float* hw2 = (const float*)d_in[19];
    const float* hb2 = (const float*)d_in[20];
    _Float16* ws = (_Float16*)d_ws;

    // weight prep: f32 [b][K][N] -> f16 [b][N][K]
    struct { int src_idx, off, K, N, b; } mats[7] = {
        { 1,  OFF_W1,  256, 128, 1 },
        { 5,  OFF_W2,  128,  64, 1 },
        { 7,  OFF_EW1,  64, 128, 4 },
        { 9,  OFF_EW2, 128,  64, 4 },
        { 11, OFF_GW1,  64,  32, 1 },
        { 15, OFF_SW,   64,  64, 1 },
        { 17, OFF_HW1,  64,  32, 3 },
    };
    for (int i = 0; i < 7; ++i) {
        int total = mats[i].K * mats[i].N * mats[i].b;
        transpose_f16_kernel<<<(total + 255) / 256, 256, 0, stream>>>(
            (const float*)d_in[mats[i].src_idx], ws + mats[i].off,
            mats[i].K, mats[i].N, total);
    }

    int B = in_sizes[0] / 256;
    int grid = B / 128;  // 128 rows per block (8 waves x 16 rows)

    (void)hipFuncSetAttribute(reinterpret_cast<const void*>(moe_fused_kernel),
                              hipFuncAttributeMaxDynamicSharedMemorySize, SMEM_BYTES);

    moe_fused_kernel<<<grid, 256, SMEM_BYTES, stream>>>(
        x, b1, lng, lnb, b2, eb1, eb2, gb1, gw2, gb2, sb, hb1, hw2, hb2, ws,
        (float*)d_out);
}
